// DocREModel_66365834657970
// MI455X (gfx1250) — compile-verified, matrix operand order checked by reading, not run on hardware
//
#include <hip/hip_runtime.h>
#include <hip/hip_fp16.h>
#include <math.h>

// ---------------- problem dims ----------------
#define B_    4
#define C_    512
#define H_    768
#define NH_   12
#define E_    42
#define M_    3
#define EMB_  768
#define SEG_  256
#define L_    97
#define P_    (E_*E_)     // 1764
#define PPAD_ 1792        // 56*32, pad P to multiple of 32 (M-tile = 32)
#define BP_   (B_*P_)     // 7056
#define BPPAD_ 7072       // 221*32, pad B*P to multiple of 32

typedef _Float16 half_t;
typedef _Float16 v16h __attribute__((ext_vector_type(16)));
typedef _Float16 v8h  __attribute__((ext_vector_type(8)));
typedef float    v8f  __attribute__((ext_vector_type(8)));

union V16U { v16h v; v8h h8[2]; };

enum { ACT_NONE = 0, ACT_TANH = 1, ACT_RELU = 2 };

__device__ __forceinline__ void load_tile(const half_t* p, V16U& u) {
    u.h8[0] = *(const v8h*)(p);
    u.h8[1] = *(const v8h*)(p + 16);
}
__device__ __forceinline__ v8f wmma16(const V16U& a, const V16U& b, v8f c) {
    return __builtin_amdgcn_wmma_f32_16x16x32_f16(false, a.v, false, b.v,
                                                  (short)0, c, false, false);
}

// ---------------- small prep kernels ----------------

// f32 [K x N] -> f16 [N x K]
__global__ void k_transpose_f16(const float* __restrict__ in, half_t* __restrict__ out,
                                int K, int N) {
    int idx = blockIdx.x * blockDim.x + threadIdx.x;
    if (idx >= K * N) return;
    int k = idx / N, n = idx - k * N;
    out[(size_t)n * K + k] = (half_t)in[idx];
}

// seq [B,C,H] f32 -> seqT [B,H,C] f16
__global__ void k_seqT(const float* __restrict__ seq, half_t* __restrict__ seqT) {
    int idx = blockIdx.x * blockDim.x + threadIdx.x;
    if (idx >= B_ * C_ * H_) return;
    int b = idx / (C_ * H_);
    int r = idx - b * C_ * H_;
    int c = r / H_, h = r - c * H_;
    seqT[((size_t)b * H_ + h) * C_ + c] = (half_t)seq[idx];
}

// entity embeddings: logsumexp over M mentions -> [B,E,H] f32
__global__ void k_entity_embs(const float* __restrict__ seq, const int* __restrict__ starts,
                              float* __restrict__ ee) {
    int idx = blockIdx.x * blockDim.x + threadIdx.x;
    if (idx >= B_ * E_ * H_) return;
    int b = idx / (E_ * H_);
    int r = idx - b * E_ * H_;
    int e = r / H_, h = r - e * H_;
    const int* sp = starts + (b * E_ + e) * M_;
    float x0 = seq[((size_t)b * C_ + (sp[0] + 1)) * H_ + h];
    float x1 = seq[((size_t)b * C_ + (sp[1] + 1)) * H_ + h];
    float x2 = seq[((size_t)b * C_ + (sp[2] + 1)) * H_ + h];
    float mx = fmaxf(x0, fmaxf(x1, x2));
    float s  = expf(x0 - mx) + expf(x1 - mx) + expf(x2 - mx);
    ee[idx] = mx + logf(s);
}

// entity attention (mean over mentions) -> [B,E,NH,C] f32
__global__ void k_entity_atts(const float* __restrict__ att, const int* __restrict__ starts,
                              float* __restrict__ ea) {
    int idx = blockIdx.x * blockDim.x + threadIdx.x;
    if (idx >= B_ * E_ * NH_ * C_) return;
    int b  = idx / (E_ * NH_ * C_);
    int r  = idx - b * E_ * NH_ * C_;
    int e  = r / (NH_ * C_);
    int r2 = r - e * NH_ * C_;
    int nh = r2 / C_, c = r2 - nh * C_;
    const int* sp = starts + (b * E_ + e) * M_;
    float s = 0.f;
    #pragma unroll
    for (int m = 0; m < M_; ++m)
        s += att[(((size_t)b * NH_ + nh) * C_ + (sp[m] + 1)) * C_ + c];
    ea[idx] = s * (1.0f / (float)M_);
}

// ht_att: product over heads, mean over NH, row-normalized; f16 out [B,PPAD,C]
__global__ void k_htatt(const float* __restrict__ ea, const int* __restrict__ hts,
                        half_t* __restrict__ out) {
    int bp = blockIdx.x;                 // B_*PPAD_
    int b  = bp / PPAD_;
    int p  = bp - b * PPAD_;
    int ti = threadIdx.x;                // 256 threads
    half_t* orow = out + (size_t)bp * C_;
    if (p >= P_) {                       // zero pad rows
        orow[ti]       = (half_t)0.f;
        orow[ti + 256] = (half_t)0.f;
        return;
    }
    int h = hts[(b * P_ + p) * 2 + 0];
    int t = hts[(b * P_ + p) * 2 + 1];
    const float* hA = ea + (size_t)(b * E_ + h) * NH_ * C_;
    const float* tA = ea + (size_t)(b * E_ + t) * NH_ * C_;
    __shared__ float sv[C_];
    __shared__ float red[256];
    float local = 0.f;
    for (int c = ti; c < C_; c += 256) {
        float s = 0.f;
        #pragma unroll
        for (int nh = 0; nh < NH_; ++nh)
            s += hA[nh * C_ + c] * tA[nh * C_ + c];
        s *= (1.0f / (float)NH_);
        sv[c] = s;
        local += s;
    }
    red[ti] = local;
    __syncthreads();
    for (int off = 128; off > 0; off >>= 1) {
        if (ti < off) red[ti] += red[ti + off];
        __syncthreads();
    }
    float inv = 1.0f / (red[0] + 1e-5f);
    for (int c = ti; c < C_; c += 256)
        orow[c] = (half_t)(sv[c] * inv);
}

// pack concat inputs [hs|rs] and [ts|rs] -> f16 [BPPAD, 1536] (pad rows zeroed)
__global__ void k_pack_hts(const float* __restrict__ ee, const float* __restrict__ rs,
                           const int* __restrict__ hts,
                           half_t* __restrict__ hsrs, half_t* __restrict__ tsrs) {
    int idx = blockIdx.x * blockDim.x + threadIdx.x;   // BPPAD_ * 1536
    if (idx >= BPPAD_ * (2 * H_)) return;
    int r = idx / (2 * H_);
    int k = idx - r * (2 * H_);
    if (r >= BP_) { hsrs[idx] = (half_t)0.f; tsrs[idx] = (half_t)0.f; return; }
    int b = r / P_, p = r - b * P_;
    int h = hts[(b * P_ + p) * 2 + 0];
    int t = hts[(b * P_ + p) * 2 + 1];
    float hv, tv;
    if (k < H_) {
        hv = ee[((size_t)b * E_ + h) * H_ + k];
        tv = ee[((size_t)b * E_ + t) * H_ + k];
    } else {
        float rv = rs[((size_t)(b * PPAD_ + p)) * H_ + (k - H_)];
        hv = rv; tv = rv;
    }
    hsrs[idx] = (half_t)hv;
    tsrs[idx] = (half_t)tv;
}

// ---------------- WMMA GEMM ----------------
// D[M,N] = act(A[M,K] * Bt[N,K]^T + bias)   (A,Bt f16; D f32 or f16)
// One wave per 32x64 output tile: 8 accumulators, 8 WMMAs per K=32 step,
// explicit double-buffered register sets over a K=64 loop body.
// Grid: (M/32, N/64, batch). K multiple of 64.
template <int ACT, bool OUT_HALF>
__global__ __launch_bounds__(32)
void k_wmma_gemm(const half_t* __restrict__ A, int lda, long long strideA,
                 const half_t* __restrict__ Bt, int ldb, long long strideB,
                 float* __restrict__ Df, half_t* __restrict__ Dh, int ldd, long long strideD,
                 const float* __restrict__ bias, int K) {
    const int lane  = threadIdx.x;
    const int row16 = lane & 15;
    const int hi    = lane >> 4;
    const int mt    = blockIdx.x;
    const int ntb   = blockIdx.y;
    const int bz    = blockIdx.z;
    A  += (long long)bz * strideA;
    Bt += (long long)bz * strideB;

    const half_t* ar0 = A + (size_t)(mt * 32 + row16) * lda + hi * 8;
    const half_t* ar1 = ar0 + (size_t)16 * lda;
    const half_t* br  = Bt + (size_t)(ntb * 64 + row16) * ldb + hi * 8;
    const size_t bstep = (size_t)16 * ldb;

    v8f acc0[4] = {};
    v8f acc1[4] = {};

    V16U aA0, aA1, bA[4];   // buffer set A
    V16U aB0, aB1, bB[4];   // buffer set B

    // preload k=0 into set A
    load_tile(ar0, aA0);
    load_tile(ar1, aA1);
    #pragma unroll
    for (int t = 0; t < 4; ++t) load_tile(br + t * bstep, bA[t]);

    for (int k0 = 0; k0 < K; k0 += 64) {
        // prefetch k0+32 into set B while computing set A
        load_tile(ar0 + k0 + 32, aB0);
        load_tile(ar1 + k0 + 32, aB1);
        #pragma unroll
        for (int t = 0; t < 4; ++t) load_tile(br + t * bstep + k0 + 32, bB[t]);

        #pragma unroll
        for (int t = 0; t < 4; ++t) {
            acc0[t] = wmma16(aA0, bA[t], acc0[t]);
            acc1[t] = wmma16(aA1, bA[t], acc1[t]);
        }

        // prefetch k0+64 into set A while computing set B
        if (k0 + 64 < K) {
            load_tile(ar0 + k0 + 64, aA0);
            load_tile(ar1 + k0 + 64, aA1);
            #pragma unroll
            for (int t = 0; t < 4; ++t) load_tile(br + t * bstep + k0 + 64, bA[t]);
        }

        #pragma unroll
        for (int t = 0; t < 4; ++t) {
            acc0[t] = wmma16(aB0, bB[t], acc0[t]);
            acc1[t] = wmma16(aB1, bB[t], acc1[t]);
        }
    }

    // epilogue: bias + activation fused into stores (32 rows x 64 cols)
    long long dbase = (long long)bz * strideD;
    #pragma unroll
    for (int t = 0; t < 4; ++t) {
        int ncol = ntb * 64 + t * 16 + row16;
        float bv = bias ? bias[ncol] : 0.f;
        #pragma unroll
        for (int g = 0; g < 8; ++g) {
            int mrow0 = mt * 32 + g + hi * 8;
            float v0 = acc0[t][g] + bv;
            float v1 = acc1[t][g] + bv;
            if (ACT == ACT_TANH)      { v0 = tanhf(v0);     v1 = tanhf(v1); }
            else if (ACT == ACT_RELU) { v0 = fmaxf(v0,0.f); v1 = fmaxf(v1,0.f); }
            long long o0 = dbase + (long long)mrow0 * ldd + ncol;
            long long o1 = o0 + (long long)16 * ldd;
            if (OUT_HALF) { Dh[o0] = (half_t)v0; Dh[o1] = (half_t)v1; }
            else          { Df[o0] = v0;         Df[o1] = v1; }
        }
    }
}

// ---------------- final logits (N=97, not 16-aligned) ----------------
__global__ void k_logits(const half_t* __restrict__ htss, const float* __restrict__ Wb,
                         const float* __restrict__ bb, float* __restrict__ out) {
    int idx = blockIdx.x * blockDim.x + threadIdx.x;   // BP_ * L_
    if (idx >= BP_ * L_) return;
    int r = idx / L_, l = idx - r * L_;
    const half_t* hr = htss + (size_t)r * SEG_;
    float acc = bb[l];
    for (int k = 0; k < SEG_; ++k)
        acc += (float)hr[k] * Wb[(size_t)k * L_ + l];
    out[idx] = acc;
}

// ---------------- launcher ----------------
extern "C" void kernel_launch(void* const* d_in, const int* in_sizes, int n_in,
                              void* d_out, int out_size, void* d_ws, size_t ws_size,
                              hipStream_t stream) {
    (void)in_sizes; (void)n_in; (void)out_size; (void)ws_size;
    const float* seq    = (const float*)d_in[0];   // [B,C,H]
    const float* att    = (const float*)d_in[1];   // [B,NH,C,C]
    const int*   starts = (const int*)  d_in[2];   // [B,E,M]
    const int*   hts    = (const int*)  d_in[3];   // [B,P,2]
    const float* W_head = (const float*)d_in[4];   // [1536,768]
    const float* b_head = (const float*)d_in[5];
    const float* W_tail = (const float*)d_in[6];
    const float* b_tail = (const float*)d_in[7];
    const float* W_rel  = (const float*)d_in[8];   // [1536,768]
    const float* b_rel  = (const float*)d_in[9];
    const float* W_seg  = (const float*)d_in[10];  // [768,256]
    const float* b_seg  = (const float*)d_in[11];
    const float* W_bil  = (const float*)d_in[12];  // [256,97]
    const float* b_bil  = (const float*)d_in[13];
    float* out = (float*)d_out;

    // workspace carve-up (256B aligned)
    char* ws = (char*)d_ws;
    size_t off = 0;
    auto alloc = [&](size_t bytes) -> void* {
        void* p = ws + off;
        off = (off + bytes + 255) & ~(size_t)255;
        return p;
    };
    half_t* WhT   = (half_t*)alloc((size_t)EMB_ * 2 * H_ * 2);       // [768 x 1536]
    half_t* WtT   = (half_t*)alloc((size_t)EMB_ * 2 * H_ * 2);
    half_t* WrT   = (half_t*)alloc((size_t)EMB_ * 2 * EMB_ * 2);     // [768 x 1536]
    half_t* WsT   = (half_t*)alloc((size_t)SEG_ * EMB_ * 2);         // [256 x 768]
    half_t* seqT  = (half_t*)alloc((size_t)B_ * H_ * C_ * 2);
    float*  ee    = (float*) alloc((size_t)B_ * E_ * H_ * 4);
    float*  ea    = (float*) alloc((size_t)B_ * E_ * NH_ * C_ * 4);
    half_t* htatt = (half_t*)alloc((size_t)B_ * PPAD_ * C_ * 2);
    // aliased regions (stream-ordered reuse):
    void* regA = alloc((size_t)B_ * PPAD_ * H_ * 4);                 // rs (f32) -> hsts (f16)
    void* regB = alloc((size_t)BPPAD_ * 2 * H_ * 2);                 // hsrs (f16) -> bl (f16)
    void* regC = alloc((size_t)BPPAD_ * 2 * H_ * 2);                 // tsrs (f16) -> htss (f16)
    float*  rs    = (float*) regA;
    half_t* hsts  = (half_t*)regA;
    half_t* hsrs  = (half_t*)regB;
    half_t* bl    = (half_t*)regB;
    half_t* tsrs  = (half_t*)regC;
    half_t* htss  = (half_t*)regC;

    const int TPB = 256;
    auto nb = [](long long n, int t) { return (unsigned)((n + t - 1) / t); };

    // weight transposes (f32 -> f16 [N x K])
    k_transpose_f16<<<nb((long long)2 * H_ * EMB_, TPB), TPB, 0, stream>>>(W_head, WhT, 2 * H_, EMB_);
    k_transpose_f16<<<nb((long long)2 * H_ * EMB_, TPB), TPB, 0, stream>>>(W_tail, WtT, 2 * H_, EMB_);
    k_transpose_f16<<<nb((long long)2 * EMB_ * EMB_, TPB), TPB, 0, stream>>>(W_rel, WrT, 2 * EMB_, EMB_);
    k_transpose_f16<<<nb((long long)EMB_ * SEG_, TPB), TPB, 0, stream>>>(W_seg, WsT, EMB_, SEG_);
    k_seqT<<<nb((long long)B_ * C_ * H_, TPB), TPB, 0, stream>>>(seq, seqT);

    // gathers / reductions
    k_entity_embs<<<nb((long long)B_ * E_ * H_, TPB), TPB, 0, stream>>>(seq, starts, ee);
    k_entity_atts<<<nb((long long)B_ * E_ * NH_ * C_, TPB), TPB, 0, stream>>>(att, starts, ea);
    k_htatt<<<B_ * PPAD_, 256, 0, stream>>>(ea, hts, htatt);

    // rs = ht_att @ seq  (batched): A [PPAD x C], Bt = seqT [H x C], D f32 [PPAD x H]
    k_wmma_gemm<ACT_NONE, false><<<dim3(PPAD_ / 32, H_ / 64, B_), 32, 0, stream>>>(
        htatt, C_, (long long)PPAD_ * C_,
        seqT, C_, (long long)H_ * C_,
        rs, nullptr, H_, (long long)PPAD_ * H_,
        nullptr, C_);

    // build [hs|rs] and [ts|rs]
    k_pack_hts<<<nb((long long)BPPAD_ * 2 * H_, TPB), TPB, 0, stream>>>(ee, rs, hts, hsrs, tsrs);

    // hs = tanh([hs|rs] @ W_head + b_head) -> f16 into hsts[:, 0:768]
    k_wmma_gemm<ACT_TANH, true><<<dim3(BPPAD_ / 32, EMB_ / 64, 1), 32, 0, stream>>>(
        hsrs, 2 * H_, 0, WhT, 2 * H_, 0,
        nullptr, hsts, 2 * EMB_, 0, b_head, 2 * H_);
    // ts = tanh([ts|rs] @ W_tail + b_tail) -> f16 into hsts[:, 768:1536]
    k_wmma_gemm<ACT_TANH, true><<<dim3(BPPAD_ / 32, EMB_ / 64, 1), 32, 0, stream>>>(
        tsrs, 2 * H_, 0, WtT, 2 * H_, 0,
        nullptr, hsts + EMB_, 2 * EMB_, 0, b_tail, 2 * H_);

    // bl = [hs|ts] @ W_rel + b_rel  -> f16 [BPPAD x 768]
    k_wmma_gemm<ACT_NONE, true><<<dim3(BPPAD_ / 32, EMB_ / 64, 1), 32, 0, stream>>>(
        hsts, 2 * EMB_, 0, WrT, 2 * EMB_, 0,
        nullptr, bl, EMB_, 0, b_rel, 2 * EMB_);

    // htss = relu(bl @ W_seg + b_seg)  (scatter/conv/gather collapses to this)
    k_wmma_gemm<ACT_RELU, true><<<dim3(BPPAD_ / 32, SEG_ / 64, 1), 32, 0, stream>>>(
        bl, EMB_, 0, WsT, EMB_, 0,
        nullptr, htss, SEG_, 0, b_seg, EMB_);

    // logits = htss @ W_bil + b_bil  (N=97 -> scalar tail kernel)
    k_logits<<<nb((long long)BP_ * L_, TPB), TPB, 0, stream>>>(htss, W_bil, b_bil, out);
}
